// S4NDLayer_18373870092729
// MI455X (gfx1250) — matrix operand order repaired and checked by reading, not compiled, not run on hardware
//
#include <hip/hip_runtime.h>

// S4ND layer for MI455X (gfx1250, wave32).
// Separable causal 2D conv as batched lower-triangular Toeplitz GEMMs on
// v_wmma_f32_16x16x32_f16. Toeplitz operands materialized in LDS once per
// block so all WMMA fragment loads are contiguous aligned LDS vector loads.
// Data planes are filled with global_load_async_to_lds_b128 (ASYNCcnt).

#define DM 256
#define NS 64
#define HH 127
#define WW 127
#define BB 4
#define LP 128   // padded 1D length (127 -> 128)

typedef __attribute__((ext_vector_type(16))) _Float16 v16h;
typedef __attribute__((ext_vector_type(8)))  _Float16 v8h;
typedef __attribute__((ext_vector_type(8)))  float    v8f;

typedef __attribute__((address_space(3))) _Float16 lds_f16;

struct cpx { float x, y; };
__device__ __forceinline__ cpx cmul(cpx a, cpx b){ return {a.x*b.x - a.y*b.y, a.x*b.y + a.y*b.x}; }
__device__ __forceinline__ cpx cadd(cpx a, cpx b){ return {a.x+b.x, a.y+b.y}; }
__device__ __forceinline__ cpx csub(cpx a, cpx b){ return {a.x-b.x, a.y-b.y}; }
__device__ __forceinline__ cpx cscale(cpx a, float s){ return {a.x*s, a.y*s}; }
__device__ __forceinline__ cpx cdiv(cpx a, cpx b){
  float inv = 1.0f/(b.x*b.x + b.y*b.y);
  return {(a.x*b.x + a.y*b.y)*inv, (a.y*b.x - a.x*b.y)*inv};
}

// Async global->LDS copy (CDNA5 Tensor/async path, tracked by ASYNCcnt).
__device__ __forceinline__ void async_ld_b128(uint32_t lds_off, const void* base,
                                              uint32_t goff) {
  asm volatile("global_load_async_to_lds_b128 %0, %1, %2"
               :: "v"(lds_off), "v"(goff), "s"(base) : "memory");
}
__device__ __forceinline__ void wait_async0() {
  asm volatile("s_wait_asynccnt 0" ::: "memory");
}
__device__ __forceinline__ uint32_t lds_addr(const _Float16* p) {
  return (uint32_t)(uintptr_t)(lds_f16*)p;
}

// ---------------------------------------------------------------------------
// Kernel 1: per (t, d): Cauchy sums + Woodbury at 127 roots of unity, then
// direct 127-point inverse DFT. Kh[t][d][0..126] f16, element 127 zero pad.
// ---------------------------------------------------------------------------
__global__ __launch_bounds__(128) void s4_kernel1d(
    const float* __restrict__ p0, const float* __restrict__ q0, const float* __restrict__ lam0,
    const float* __restrict__ B0, const float* __restrict__ Ct0, const float* __restrict__ ls0,
    const float* __restrict__ p1, const float* __restrict__ q1, const float* __restrict__ lam1,
    const float* __restrict__ B1, const float* __restrict__ Ct1, const float* __restrict__ ls1,
    _Float16* __restrict__ Kh)
{
  const int t = blockIdx.x >> 8;
  const int d = blockIdx.x & 255;
  const float* p  = t ? p1   : p0;
  const float* q  = t ? q1   : q0;
  const float* lm = t ? lam1 : lam0;
  const float* Bm = t ? B1   : B0;
  const float* Ct = t ? Ct1  : Ct0;
  const float* ls = t ? ls1  : ls0;

  __shared__ float atre[LP], atim[LP];

  const int l = threadIdx.x;
  if (l < HH) {
    float ang = 6.28318530717958647692f * (float)l / 127.0f;
    float sw, cw; sincosf(ang, &sw, &cw);                 // omega = cw + i*sw
    cpx one_m = {1.f - cw, -sw};
    cpx one_p = {1.f + cw,  sw};
    float dp = one_p.x*one_p.x + one_p.y*one_p.y;
    cpx ratio = cscale(cmul(one_m, cpx{one_p.x, -one_p.y}), 1.f/dp);
    float step = __expf(ls[d]);
    cpx g = cscale(ratio, 2.f/step);                      // bilinear transform
    cpx c = { 2.f*one_p.x/dp, -2.f*one_p.y/dp };          // 2/(1+omega)
    cpx k00={0,0}, k01={0,0}, k10={0,0}, k11={0,0};
    for (int n = 0; n < NS; ++n) {
      cpx pn = {p[2*n],   p[2*n+1]};
      cpx qc = {q[2*n],  -q[2*n+1]};
      cpx la = {lm[2*n],  lm[2*n+1]};
      cpx Bn = {Bm[(d*NS+n)*2],  Bm[(d*NS+n)*2+1]};
      cpx Cc = {Ct[(d*NS+n)*2], -Ct[(d*NS+n)*2+1]};
      cpx den = csub(g, la);
      float inv = 1.f/(den.x*den.x + den.y*den.y);
      cpx r = {den.x*inv, -den.y*inv};
      k00 = cadd(k00, cmul(cmul(Cc, Bn), r));
      k01 = cadd(k01, cmul(cmul(Cc, pn), r));
      k10 = cadd(k10, cmul(cmul(qc, Bn), r));
      k11 = cadd(k11, cmul(cmul(qc, pn), r));
    }
    cpx corr = cdiv(cmul(k01, k10), cpx{1.f + k11.x, k11.y});
    cpx at = cmul(c, csub(k00, corr));
    atre[l] = at.x; atim[l] = at.y;
  }
  __syncthreads();

  const int k = threadIdx.x;
  _Float16* out = Kh + ((size_t)t*DM + d) * LP;
  if (k < HH) {
    float base = 6.28318530717958647692f * (float)k / 127.0f;
    float acc = 0.f;
    for (int ll = 0; ll < HH; ++ll) {
      float s, cc; sincosf(base*(float)ll, &s, &cc);
      acc += atre[ll]*cc + atim[ll]*s;
    }
    out[k] = (_Float16)(acc * (1.0f/127.0f));
  } else {
    out[k] = (_Float16)0.f;
  }
}

// ---------------------------------------------------------------------------
// Pack: u(B,H,W,D) f32 -> 1024 zero-padded 128x128 f16 planes in [j][i]
// layout (so conv_h's LDS fill is a straight async copy). Coalesced via LDS.
// ---------------------------------------------------------------------------
__global__ __launch_bounds__(256) void s4_pack(const float* __restrict__ u,
                                               _Float16* __restrict__ uth)
{
  const int b  = blockIdx.x >> 7;
  const int j  = blockIdx.x & 127;        // includes pad row j==127
  const int i0 = blockIdx.y * 32;
  const int d0 = blockIdx.z * 32;
  __shared__ float t[32][33];
  const int tx = threadIdx.x, ty = threadIdx.y;
  #pragma unroll
  for (int r = 0; r < 4; ++r) {
    int il = ty + 8*r, i = i0 + il;
    float v = 0.f;
    if (i < HH && j < WW)
      v = u[(((size_t)b*HH + i)*WW + j)*DM + d0 + tx];
    t[il][tx] = v;
  }
  __syncthreads();
  #pragma unroll
  for (int r = 0; r < 4; ++r) {
    int dl = ty + 8*r;
    uth[((size_t)(b*DM + d0 + dl))*(LP*LP) + j*LP + i0 + tx] = (_Float16)t[tx][dl];
  }
}

// ---------------------------------------------------------------------------
// Pass 1 (conv along H): per plane (b,d): tmp[i][j] = sum_a K0[d,i-a] u[a][j].
// LDS: tz = row-major Toeplitz T0[m][k]=K0[m-k]; up = data plane [j][a]
// (async-filled). A-fragment: two 16B v8h loads from tz row m; B-fragment:
// one 32B v16h load from up row j. Triangular skip kb <= w/2.
// ---------------------------------------------------------------------------
__global__ __launch_bounds__(256) void s4_conv_h(const _Float16* __restrict__ Kh,
                                                 const _Float16* __restrict__ uth,
                                                 _Float16* __restrict__ tmph)
{
  const int plane = blockIdx.x;
  const int d = plane & 255;
  __shared__ __align__(32) _Float16 up[LP*LP];   // [j][a]
  __shared__ __align__(32) _Float16 tz[LP*LP];   // Toeplitz [m][k]
  const int tid = threadIdx.x;

  // Async fill of the data plane (32 KB), bypassing VGPRs.
  const _Float16* src = uth + (size_t)plane*(LP*LP);
  {
    uint32_t ub = lds_addr(up);
    #pragma unroll
    for (int r = 0; r < 8; ++r) {
      uint32_t off = (uint32_t)(tid + 256*r) * 16u;   // 16 B per lane
      async_ld_b128(ub + off, src, off);
    }
  }
  // Build Toeplitz operand from the (L2-resident) K0 vector.
  const _Float16* kg = Kh + (size_t)d*LP;
  #pragma unroll 4
  for (int r = 0; r < (LP*LP)/256; ++r) {
    int idx = tid + 256*r;
    int mm = idx >> 7, kk = idx & 127;
    tz[idx] = (mm >= kk) ? kg[mm - kk] : (_Float16)0.f;
  }
  wait_async0();
  __syncthreads();

  const int w = tid >> 5, lane = tid & 31, half = lane >> 4, lr = lane & 15;
  const int mi = w*16, m = mi + lr;
  const v8f zero = {};
  v8f acc[8];
  #pragma unroll
  for (int nt = 0; nt < 8; ++nt) acc[nt] = zero;

  const int kbmax = w >> 1;                       // lower-triangular skip
  for (int kb = 0; kb <= kbmax; ++kb) {
    const int koff = kb*32;
    v8h alo = *(const v8h*)(&tz[m*LP + koff + half*16]);       // e<8 : K=koff+half*16+e
    v8h ahi = *(const v8h*)(&tz[m*LP + koff + half*16 + 8]);   // e>=8: K=koff+half*16+e
    v16h a = __builtin_shufflevector(alo, ahi, 0,1,2,3,4,5,6,7,8,9,10,11,12,13,14,15);
    #pragma unroll
    for (int nt = 0; nt < 8; ++nt) {
      int col = nt*16 + lr;
      v16h b = *(const v16h*)(&up[col*LP + koff + half*16]);   // K = half*16 + e
      acc[nt] = __builtin_amdgcn_wmma_f32_16x16x32_f16(false, a, false, b,
                                                       (short)0, acc[nt], false, false);
    }
  }

  _Float16* dst = tmph + (size_t)plane*(LP*LP);
  #pragma unroll
  for (int nt = 0; nt < 8; ++nt)
    #pragma unroll
    for (int r = 0; r < 8; ++r)
      dst[(mi + r + 8*half)*LP + nt*16 + lr] = (_Float16)acc[nt][r];
}

// ---------------------------------------------------------------------------
// Pass 2 (conv along W): out[i][j] = sum_a tmp[i][a] K1[d,j-a].
// LDS: tp = tmp plane [i][a] (async-filled); tz = column-major Toeplitz'
// tz[j][a]=K1[j-a]. A: two v8h loads from tp row i; B: one v16h load from
// tz row j. Triangular skip nt >= 2*kb.
// ---------------------------------------------------------------------------
__global__ __launch_bounds__(256) void s4_conv_w(const _Float16* __restrict__ Kh,
                                                 const _Float16* __restrict__ tmph,
                                                 _Float16* __restrict__ outh)
{
  const int plane = blockIdx.x;
  const int d = plane & 255;
  __shared__ __align__(32) _Float16 tp[LP*LP];   // [i][a]
  __shared__ __align__(32) _Float16 tz[LP*LP];   // Toeplitz' [j][a]
  const int tid = threadIdx.x;

  const _Float16* src = tmph + (size_t)plane*(LP*LP);
  {
    uint32_t ub = lds_addr(tp);
    #pragma unroll
    for (int r = 0; r < 8; ++r) {
      uint32_t off = (uint32_t)(tid + 256*r) * 16u;
      async_ld_b128(ub + off, src, off);
    }
  }
  const _Float16* kg = Kh + (size_t)(DM + d)*LP;
  #pragma unroll 4
  for (int r = 0; r < (LP*LP)/256; ++r) {
    int idx = tid + 256*r;
    int jj = idx >> 7, aa = idx & 127;
    tz[idx] = (jj >= aa) ? kg[jj - aa] : (_Float16)0.f;
  }
  wait_async0();
  __syncthreads();

  const int w = tid >> 5, lane = tid & 31, half = lane >> 4, lr = lane & 15;
  const int mi = w*16, m = mi + lr;
  const v8f zero = {};
  v8f acc[8];
  #pragma unroll
  for (int nt = 0; nt < 8; ++nt) acc[nt] = zero;

  #pragma unroll
  for (int kb = 0; kb < 4; ++kb) {
    const int koff = kb*32;
    v8h alo = *(const v8h*)(&tp[m*LP + koff + half*16]);
    v8h ahi = *(const v8h*)(&tp[m*LP + koff + half*16 + 8]);
    v16h a = __builtin_shufflevector(alo, ahi, 0,1,2,3,4,5,6,7,8,9,10,11,12,13,14,15);
    for (int nt = 2*kb; nt < 8; ++nt) {          // B zero when koff > nj+15
      int col = nt*16 + lr;
      v16h b = *(const v16h*)(&tz[col*LP + koff + half*16]);
      acc[nt] = __builtin_amdgcn_wmma_f32_16x16x32_f16(false, a, false, b,
                                                       (short)0, acc[nt], false, false);
    }
  }

  _Float16* dst = outh + (size_t)plane*(LP*LP);
  #pragma unroll
  for (int nt = 0; nt < 8; ++nt)
    #pragma unroll
    for (int r = 0; r < 8; ++r)
      dst[(mi + r + 8*half)*LP + nt*16 + lr] = (_Float16)acc[nt][r];
}

// ---------------------------------------------------------------------------
// Unpack: [i][j] planes f16 -> y(B,H,W,D) f32, fused + Dp[d]*u.
// ---------------------------------------------------------------------------
__global__ __launch_bounds__(256) void s4_unpack(const _Float16* __restrict__ outh,
                                                 const float* __restrict__ u,
                                                 const float* __restrict__ Dp,
                                                 float* __restrict__ y)
{
  const int b  = blockIdx.x / HH;
  const int i  = blockIdx.x % HH;
  const int j0 = blockIdx.y * 32;
  const int d0 = blockIdx.z * 32;
  __shared__ float t[32][33];
  const int tx = threadIdx.x, ty = threadIdx.y;
  #pragma unroll
  for (int r = 0; r < 4; ++r) {
    int dl = ty + 8*r;
    t[tx][dl] = (float)outh[((size_t)(b*DM + d0 + dl))*(LP*LP) + i*LP + j0 + tx];
  }
  __syncthreads();
  #pragma unroll
  for (int r = 0; r < 4; ++r) {
    int jl = ty + 8*r, j = j0 + jl;
    if (j < WW) {
      size_t idx = (((size_t)b*HH + i)*WW + j)*DM + d0 + tx;
      y[idx] = t[jl][tx] + Dp[d0 + tx] * u[idx];
    }
  }
}

// ---------------------------------------------------------------------------
extern "C" void kernel_launch(void* const* d_in, const int* in_sizes, int n_in,
                              void* d_out, int out_size, void* d_ws, size_t ws_size,
                              hipStream_t stream) {
  (void)in_sizes; (void)n_in; (void)out_size; (void)ws_size;
  const float* u    = (const float*)d_in[0];
  const float* p0   = (const float*)d_in[1];
  const float* q0   = (const float*)d_in[2];
  const float* lam0 = (const float*)d_in[3];
  const float* B0   = (const float*)d_in[4];
  const float* Ct0  = (const float*)d_in[5];
  const float* ls0  = (const float*)d_in[6];
  const float* p1   = (const float*)d_in[7];
  const float* q1   = (const float*)d_in[8];
  const float* lam1 = (const float*)d_in[9];
  const float* B1   = (const float*)d_in[10];
  const float* Ct1  = (const float*)d_in[11];
  const float* ls1  = (const float*)d_in[12];
  const float* Dp   = (const float*)d_in[13];
  float* y = (float*)d_out;

  char* ws = (char*)d_ws;
  _Float16* Kh   = (_Float16*)ws;                                       // 128 KB
  _Float16* uth  = (_Float16*)(ws + (size_t)131072);                    // 32 MB
  _Float16* tmph = (_Float16*)(ws + (size_t)131072 + (size_t)33554432); // 32 MB
  _Float16* outh = uth;   // pass-2 output aliases the consumed packed-u buffer

  s4_kernel1d<<<512, 128, 0, stream>>>(p0,q0,lam0,B0,Ct0,ls0,
                                       p1,q1,lam1,B1,Ct1,ls1, Kh);
  s4_pack  <<<dim3(BB*LP, 4, 8), dim3(32, 8), 0, stream>>>(u, uth);
  s4_conv_h<<<BB*DM, 256, 0, stream>>>(Kh, uth, tmph);
  s4_conv_w<<<BB*DM, 256, 0, stream>>>(Kh, tmph, outh);
  s4_unpack<<<dim3(BB*HH, 4, 8), dim3(32, 8), 0, stream>>>(outh, u, Dp, y);
}
